// SpectralGatedDeltaNet_23854248362223
// MI455X (gfx1250) — compile-verified, matrix operand order checked by reading, not run on hardware
//
#include <hip/hip_runtime.h>
#include <hip/hip_bf16.h>
#include <math.h>

// Problem constants (from reference)
#define BB 2
#define TT 128
#define DD 1024
#define HH 8
#define RR 8
#define DK 64
#define DV 64
#define MROWS (BB*TT)          // 256

typedef __attribute__((ext_vector_type(16))) _Float16 v16h;
typedef __attribute__((ext_vector_type(8)))  _Float16 v8h;
typedef __attribute__((ext_vector_type(8)))  float    v8f;

__device__ __forceinline__ float sigmoid_fast(float x) {
    return __builtin_amdgcn_rcpf(1.0f + __expf(-x));
}

// ---------------------------------------------------------------------------
// WMMA GEMM with native-layout f16 operands.
//   A  : [M,K] f16 row-major
//   Bt : [N,K] f16 row-major (i.e. W[K,N] pre-transposed)
//   C  : [M,N] f32 row-major
// Per ISA 7.12.2 the 16x32 f16 A fragment for lane L (m=L&15, hi=L>>4) is
//   halves 0..7  = K  8*hi + 0..7   (contiguous)
//   halves 8..15 = K 16+8*hi + 0..7 (contiguous)
// so each fragment = two global_load_b128 per lane; same for B against Bt.
// 8 waves / block, one 16x16 C tile per wave (linear tile id).
// K must be a multiple of 64 (manual 2x unroll -> 2 WMMAs/iter).
// ---------------------------------------------------------------------------
__global__ void gemm16_wmma(const _Float16* __restrict__ A,
                            const _Float16* __restrict__ Bt,
                            float* __restrict__ C,
                            int M, int N, int K, int act)
{
    const int lane = threadIdx.x & 31;
    const int wid  = threadIdx.x >> 5;
    const int tid  = blockIdx.x * 8 + wid;          // linear 16x16 tile id
    const int ntiles = N >> 4;
    const int m0 = (tid / ntiles) << 4;
    const int n0 = (tid % ntiles) << 4;
    if (m0 >= M) return;                            // wave-uniform guard

    const int hi = lane >> 4;
    const int lo = lane & 15;
    const _Float16* arow = A  + (size_t)(m0 + lo) * K + 8*hi;
    const _Float16* brow = Bt + (size_t)(n0 + lo) * K + 8*hi;

    v8f acc = {};
    for (int k0 = 0; k0 < K; k0 += 64) {
        v8h al0 = *(const v8h*)(arow + k0);
        v8h ah0 = *(const v8h*)(arow + k0 + 16);
        v8h bl0 = *(const v8h*)(brow + k0);
        v8h bh0 = *(const v8h*)(brow + k0 + 16);
        v8h al1 = *(const v8h*)(arow + k0 + 32);
        v8h ah1 = *(const v8h*)(arow + k0 + 48);
        v8h bl1 = *(const v8h*)(brow + k0 + 32);
        v8h bh1 = *(const v8h*)(brow + k0 + 48);
        v16h a0 = __builtin_shufflevector(al0, ah0, 0,1,2,3,4,5,6,7,8,9,10,11,12,13,14,15);
        v16h b0 = __builtin_shufflevector(bl0, bh0, 0,1,2,3,4,5,6,7,8,9,10,11,12,13,14,15);
        v16h a1 = __builtin_shufflevector(al1, ah1, 0,1,2,3,4,5,6,7,8,9,10,11,12,13,14,15);
        v16h b1 = __builtin_shufflevector(bl1, bh1, 0,1,2,3,4,5,6,7,8,9,10,11,12,13,14,15);
        acc = __builtin_amdgcn_wmma_f32_16x16x32_f16(false, a0, false, b0,
                                                     (short)0, acc, false, false);
        acc = __builtin_amdgcn_wmma_f32_16x16x32_f16(false, a1, false, b1,
                                                     (short)0, acc, false, false);
    }
#pragma unroll
    for (int r = 0; r < 8; ++r) {
        float x = acc[r];
        if (act == 1) x = sigmoid_fast(x);
        C[(size_t)(m0 + r + 8*hi) * N + (n0 + lo)] = x;
    }
}

// ---------------------------------------------------------------------------
// One-time layout prep: f32 -> f16 copy, and f32 [K,N] -> f16 [N,K] transpose.
// ---------------------------------------------------------------------------
__global__ void cvt_f16_kernel(const float* __restrict__ in,
                               _Float16* __restrict__ out, int n)
{
    const int i = blockIdx.x * blockDim.x + threadIdx.x;
    if (i < n) out[i] = (_Float16)in[i];
}

__global__ void transpose_f16_kernel(const float* __restrict__ in,   // [K,N]
                                     _Float16* __restrict__ out,     // [N,K]
                                     int K, int N)
{
    const int i = blockIdx.x * blockDim.x + threadIdx.x;
    if (i >= K * N) return;
    const int kk = i % K;
    const int nn = i / K;
    out[i] = (_Float16)in[(size_t)kk * N + nn];      // coalesced writes
}

// ---------------------------------------------------------------------------
// Precompute per-(h,r) spectral parameters.
// ---------------------------------------------------------------------------
__global__ void params_kernel(const float* __restrict__ mode_logits,
                              const float* __restrict__ log_decay,
                              const float* __restrict__ omega_log_scale,
                              float* __restrict__ cosw,
                              float* __restrict__ sinw,
                              float* __restrict__ rho,
                              float* __restrict__ modew)
{
    const int t = threadIdx.x;      // 0..63
    if (t >= HH*RR) return;
    const int h = t >> 3;

    const float ld = log_decay[t];
    const float sp = (ld > 20.0f) ? ld : log1pf(expf(ld));   // softplus
    rho[t] = expf(-sp);

    float mx = -1e30f;
    for (int j = 0; j < RR; ++j) mx = fmaxf(mx, mode_logits[h*RR + j]);
    float sum = 0.0f;
    for (int j = 0; j < RR; ++j) sum += expf(mode_logits[h*RR + j] - mx);
    modew[t] = expf(mode_logits[t] - mx) / sum;

    const float os = expf(omega_log_scale[t]);
    for (int d = 0; d < DK; ++d) {
        const float pidx = (float)(d & ~1);                  // repeat_interleave
        const float invf = powf(10000.0f, -pidx / (float)DK);
        const float w = os * invf;
        cosw[t*DK + d] = cosf(w);
        sinw[t*DK + d] = sinf(w);
    }
}

// ---------------------------------------------------------------------------
// Recurrent scan: one wave32 per independent state row (b,h,r,v); 2 Dk-state
// elements per lane in registers; two butterfly wave reductions per step.
// ys layout: [B*T, H, R, DV].
// ---------------------------------------------------------------------------
__global__ void scan_kernel(const float* __restrict__ q,
                            const float* __restrict__ k,
                            const float* __restrict__ v,
                            const float* __restrict__ beta,
                            const float* __restrict__ tg,
                            const float* __restrict__ cosw,
                            const float* __restrict__ sinw,
                            const float* __restrict__ rho,
                            const float* __restrict__ modew,
                            float* __restrict__ ys)
{
    const int gw   = blockIdx.x * (blockDim.x >> 5) + (threadIdx.x >> 5);
    const int lane = threadIdx.x & 31;
    const int vrow = gw & 63;
    const int r    = (gw >> 6) & 7;
    const int h    = (gw >> 9) & 7;
    const int b    = (gw >> 12) & 1;
    const int hr   = h*RR + r;

    const float c0 = cosw[hr*DK + 2*lane];
    const float c1 = cosw[hr*DK + 2*lane + 1];
    const float s0 = sinw[hr*DK + 2*lane];
    const float s1 = sinw[hr*DK + 2*lane + 1];
    const float rho_hr = rho[hr];
    const float mw     = modew[hr];

    const float2* k2 = (const float2*)k;
    const float2* q2 = (const float2*)q;

    float sr0 = 0.f, sr1 = 0.f, si0 = 0.f, si1 = 0.f;

    for (int t = 0; t < TT; ++t) {
        const int bt = b*TT + t;
        const int bh = bt*HH + h;                 // flat (b,t,h)
        const float2 kt = k2[bh*32 + lane];
        const float2 qt = q2[bh*32 + lane];
        const float tgt = tg[bh*RR + r];
        const float bet = beta[bh*RR + r];
        const float vt  = v[bh*DV + vrow];

        const float decay = tgt * rho_hr;
        const float rr0 = decay * (c0*sr0 - s0*si0);
        const float rr1 = decay * (c1*sr1 - s1*si1);
        const float ri0 = decay * (s0*sr0 + c0*si0);
        const float ri1 = decay * (s1*sr1 + c1*si1);

        float pred = rr0*kt.x + rr1*kt.y;
#pragma unroll
        for (int off = 16; off; off >>= 1) pred += __shfl_xor(pred, off, 32);

        const float bw = bet * (vt - pred);
        sr0 = rr0 + bw*kt.x;
        sr1 = rr1 + bw*kt.y;
        si0 = ri0;
        si1 = ri1;

        float rd = sr0*qt.x + sr1*qt.y;
#pragma unroll
        for (int off = 16; off; off >>= 1) rd += __shfl_xor(rd, off, 32);

        if (lane == 0)
            ys[(size_t)(bh*RR + r)*DV + vrow] = rd * mw;
    }
}

// ---------------------------------------------------------------------------
// ogh[bt, h*DV+v] = f16( gate * sum_r ys[bt,h,r,v] )  (feeds the final WMMA)
// ---------------------------------------------------------------------------
__global__ void combine_kernel(const float* __restrict__ ys,
                               const float* __restrict__ g,
                               _Float16* __restrict__ ogh)
{
    const int i = blockIdx.x * blockDim.x + threadIdx.x;   // 0..MROWS*H*DV-1
    const int vv = i & 63;
    const int h  = (i >> 6) & 7;
    const int bt = i >> 9;
    const size_t base = (size_t)((bt*HH + h)*RR)*DV + vv;
    float s = 0.0f;
#pragma unroll
    for (int r = 0; r < RR; ++r) s += ys[base + r*DV];
    ogh[i] = (_Float16)(s * g[i]);
}

// ---------------------------------------------------------------------------
extern "C" void kernel_launch(void* const* d_in, const int* in_sizes, int n_in,
                              void* d_out, int out_size, void* d_ws, size_t ws_size,
                              hipStream_t stream)
{
    (void)in_sizes; (void)n_in; (void)out_size; (void)ws_size;
    const float* x    = (const float*)d_in[0];
    const float* Wq   = (const float*)d_in[1];
    const float* Wk   = (const float*)d_in[2];
    const float* Wv   = (const float*)d_in[3];
    const float* Wb   = (const float*)d_in[4];
    const float* Wtg  = (const float*)d_in[5];
    const float* mlg  = (const float*)d_in[6];
    const float* ldc  = (const float*)d_in[7];
    const float* oms  = (const float*)d_in[8];
    const float* Wg   = (const float*)d_in[9];
    const float* Wo   = (const float*)d_in[10];

    // ---- f32 scratch ----
    float* ws    = (float*)d_ws;
    float* q     = ws;                        // 256*512
    float* k     = q    + MROWS*512;
    float* v     = k    + MROWS*512;
    float* g     = v    + MROWS*512;
    float* beta  = g    + MROWS*512;          // 256*64
    float* tg    = beta + MROWS*64;
    float* cosw  = tg   + MROWS*64;           // 64*64
    float* sinw  = cosw + HH*RR*DK;
    float* rho   = sinw + HH*RR*DK;           // 64
    float* modew = rho  + HH*RR;
    float* ys    = modew + HH*RR;             // 256*8*8*64 = 1048576
    float* fend  = ys   + MROWS*HH*RR*DV;

    // ---- f16 scratch (16B-aligned: float count above is a multiple of 4) ----
    _Float16* hb   = (_Float16*)fend;
    _Float16* xh   = hb;                      // 256*1024
    _Float16* Wqt  = xh   + MROWS*DD;         // 512*1024 each
    _Float16* Wkt  = Wqt  + 512*DD;
    _Float16* Wvt  = Wkt  + 512*DD;
    _Float16* Wgt  = Wvt  + 512*DD;
    _Float16* Wbt  = Wgt  + 512*DD;           // 64*1024 each
    _Float16* Wtgt = Wbt  + 64*DD;
    _Float16* Wot  = Wtgt + 64*DD;            // 1024*512
    _Float16* ogh  = Wot  + DD*512;           // 256*512

    params_kernel<<<1, 64, 0, stream>>>(mlg, ldc, oms, cosw, sinw, rho, modew);

    // layout prep
    cvt_f16_kernel<<<(MROWS*DD)/256, 256, 0, stream>>>(x, xh, MROWS*DD);
    transpose_f16_kernel<<<(DD*512)/256, 256, 0, stream>>>(Wq,  Wqt,  DD, 512);
    transpose_f16_kernel<<<(DD*512)/256, 256, 0, stream>>>(Wk,  Wkt,  DD, 512);
    transpose_f16_kernel<<<(DD*512)/256, 256, 0, stream>>>(Wv,  Wvt,  DD, 512);
    transpose_f16_kernel<<<(DD*512)/256, 256, 0, stream>>>(Wg,  Wgt,  DD, 512);
    transpose_f16_kernel<<<(DD*64)/256,  256, 0, stream>>>(Wb,  Wbt,  DD, 64);
    transpose_f16_kernel<<<(DD*64)/256,  256, 0, stream>>>(Wtg, Wtgt, DD, 64);
    transpose_f16_kernel<<<(512*DD)/256, 256, 0, stream>>>(Wo,  Wot,  512, DD);

    // projections: tiles = (M/16)*(N/16); 8 tiles (waves) per 256-thread block
    gemm16_wmma<<<(16*32)/8, 256, 0, stream>>>(xh, Wqt,  q,    MROWS, 512, DD, 0);
    gemm16_wmma<<<(16*32)/8, 256, 0, stream>>>(xh, Wkt,  k,    MROWS, 512, DD, 0);
    gemm16_wmma<<<(16*32)/8, 256, 0, stream>>>(xh, Wvt,  v,    MROWS, 512, DD, 0);
    gemm16_wmma<<<(16*4)/8,  256, 0, stream>>>(xh, Wbt,  beta, MROWS, 64,  DD, 1);
    gemm16_wmma<<<(16*4)/8,  256, 0, stream>>>(xh, Wtgt, tg,   MROWS, 64,  DD, 1);
    gemm16_wmma<<<(16*32)/8, 256, 0, stream>>>(xh, Wgt,  g,    MROWS, 512, DD, 1);

    // 8192 rows, 1 wave each, 8 waves per 256-thread block -> 1024 blocks
    scan_kernel<<<1024, 256, 0, stream>>>(q, k, v, beta, tg, cosw, sinw, rho, modew, ys);

    combine_kernel<<<(MROWS*HH*DV)/256, 256, 0, stream>>>(ys, g, ogh);

    gemm16_wmma<<<(16*64)/8, 256, 0, stream>>>(ogh, Wot, (float*)d_out,
                                               MROWS, DD, 512, 0);
}
